// DCRNNModel_618475291217
// MI455X (gfx1250) — matrix operand order repaired and checked by reading, not compiled, not run on hardware
//
#include <hip/hip_runtime.h>
#include <hip/hip_bf16.h>

typedef __attribute__((ext_vector_type(16))) _Float16 v16h;
typedef __attribute__((ext_vector_type(8)))  float    v8f;

#define N_FEAT  32
#define BN_EPS  1e-5f

// ---------------------------------------------------------------------------
// Weight prep: build f16 weight matrices for the WMMA GEMMs.
// Wgate [160 x 128]: rows = K blocks {x, Txo1, Txi1, Txo2, Txi2} (32 each),
// cols 0..63 = Z gate, 64..127 = H gate. Only first 32 rows of fin matter
// (H0 == 0). biascat[128] = [bz | bh].
// ---------------------------------------------------------------------------
__global__ void prep_weights_kernel(const float* __restrict__ Wz,
                                    const float* __restrict__ Wh,
                                    const float* __restrict__ bz,
                                    const float* __restrict__ bh,
                                    const float* __restrict__ gcn_w,
                                    _Float16* __restrict__ Wgate,
                                    _Float16* __restrict__ Wgcn,
                                    float* __restrict__ biascat) {
    int t = blockIdx.x * blockDim.x + threadIdx.x;
    if (t < 160 * 128) {
        int krow = t >> 7;          // 0..159
        int n    = t & 127;         // 0..127
        int g    = n >> 6;          // 0 = z, 1 = h
        int nn   = n & 63;
        int kb   = krow >> 5;       // 0..4
        int kk   = krow & 31;
        const float* W = g ? Wh : Wz;   // [2,3,96,64]
        float v;
        if (kb == 0) {
            v = W[((0 * 3 + 0) * 96 + kk) * 64 + nn]
              + W[((1 * 3 + 0) * 96 + kk) * 64 + nn];
        } else {
            int d = (kb == 2 || kb == 4) ? 1 : 0;  // out vs in direction
            int k = (kb <= 2) ? 1 : 2;             // hop
            v = W[((d * 3 + k) * 96 + kk) * 64 + nn];
        }
        Wgate[krow * 128 + n] = (_Float16)v;
    }
    if (t < 64 * 64) Wgcn[t] = (_Float16)gcn_w[t];
    if (t < 128)     biascat[t] = (t < 64) ? bz[t] : bh[t - 64];
}

// ---------------------------------------------------------------------------
// Weighted degrees (atomics) + unweighted in-degree count for the GCN norm.
// ---------------------------------------------------------------------------
__global__ void degree_kernel(const int* __restrict__ row, const int* __restrict__ col,
                              const float* __restrict__ ew,
                              float* __restrict__ deg_out, float* __restrict__ deg_in,
                              float* __restrict__ dcnt, int E) {
    int e = blockIdx.x * blockDim.x + threadIdx.x;
    if (e >= E) return;
    float w = ew[e];
    atomicAdd(&deg_out[row[e]], w);
    atomicAdd(&deg_in[col[e]], w);
    atomicAdd(&dcnt[col[e]], 1.0f);
}

__global__ void invert_kernel(float* __restrict__ deg_out, float* __restrict__ deg_in,
                              float* __restrict__ dcnt, float* __restrict__ dis, int n) {
    int i = blockIdx.x * blockDim.x + threadIdx.x;
    if (i >= n) return;
    float a = deg_out[i]; deg_out[i] = (a > 0.0f) ? (1.0f / a) : 0.0f;
    float b = deg_in[i];  deg_in[i]  = (b > 0.0f) ? (1.0f / b) : 0.0f;
    dis[i] = rsqrtf(dcnt[i] + 1.0f);   // + self loop, always > 0
}

// ---------------------------------------------------------------------------
// One diffusion hop, both directions at once. Lane == feature (32 feats),
// so each wave's atomics coalesce into one 128B L2 atomic per destination row.
// ---------------------------------------------------------------------------
__global__ void prop_kernel(const float* __restrict__ src_o, const float* __restrict__ src_i,
                            float* __restrict__ dst_o, float* __restrict__ dst_i,
                            const int* __restrict__ row, const int* __restrict__ col,
                            const float* __restrict__ ew,
                            const float* __restrict__ do_inv, const float* __restrict__ di_inv,
                            int E) {
    int gid = blockIdx.x * blockDim.x + threadIdx.x;
    int e = gid >> 5;
    int f = gid & 31;
    if (e >= E) return;
    int r = row[e], c = col[e];
    float w = ew[e];
    float wo = w * do_inv[r];
    float wi = w * di_inv[c];
    atomicAdd(&dst_o[c * 32 + f], wo * src_o[r * 32 + f]);
    atomicAdd(&dst_i[r * 32 + f], wi * src_i[c * 32 + f]);
}

// ---------------------------------------------------------------------------
// Templated WMMA GEMM: out[nrows, NCOLS] = concat_k(src[kb]) @ W + bias.
// K consumed in NKB blocks of 32 from f32 sources (cvt f16 on the fly per the
// 16-bit A-fragment layout: lanes 0-15 K{0..7,16..23}, lanes 16-31 K{8..15,
// 24..31}). W is staged in LDS pre-swizzled into B-fragment order so each
// lane reads one contiguous 32B v16h (ds_load_b128 x2). Fully unrolled K ->
// NKB back-to-back v_wmma_f32_16x16x32_f16 per wave tile.
// ---------------------------------------------------------------------------
struct GemmArgs {
    const float* src[5];
    int          ld[5];
    const _Float16* W;     // [NKB*32, NCOLS] row-major
    const float* bias;     // [NCOLS] or nullptr
    float*       out;      // [nrows, NCOLS]
    int          nrows;    // multiple of 16 (50000 = 3125*16)
};

template <int NKB, int NCOLS>
__global__ void __launch_bounds__(256) gemm_wmma_kernel(GemmArgs ga) {
    constexpr int WPRT  = NCOLS / 16;      // waves per M-tile (8 or 4)
    constexpr int TPB   = 8 / WPRT;        // M-tiles per block
    constexpr int TOTAL = NKB * 32 * NCOLS;
    __shared__ _Float16 sW[TOTAL];

    int tid = threadIdx.x;
    // cooperative LDS fill, swizzled to B-fragment order:
    //   sW[((kb*WPRT + ntile)*32 + lane)*16 + j]
    //     = W[(kb*32 + (lane>>4)*8 + (j&7) + (j>>3)*16) * NCOLS + ntile*16 + (lane&15)]
    for (int idx = tid; idx < TOTAL; idx += 256) {
        int kb    = idx / (WPRT * 512);
        int rem   = idx - kb * (WPRT * 512);
        int ntile = rem >> 9;
        int li    = rem & 511;
        int lane  = li >> 4;
        int j     = li & 15;
        int k     = kb * 32 + ((lane >> 4) << 3) + (j & 7) + ((j >> 3) << 4);
        int n     = (ntile << 4) + (lane & 15);
        sW[idx] = ga.W[k * NCOLS + n];
    }
    __syncthreads();

    int wave  = tid >> 5;
    int lane  = tid & 31;
    int mtile = blockIdx.x * TPB + wave / WPRT;
    int ntile = wave % WPRT;
    if (mtile >= (ga.nrows >> 4)) return;     // wave-uniform guard

    int m     = (mtile << 4) + (lane & 15);
    int khalf = (lane >> 4) * 8;              // A fragment K split
    const v16h* bfrag = (const v16h*)&sW[(((wave % WPRT) * 32 + lane) << 4)];

    v8f acc = {};
#pragma unroll
    for (int kb = 0; kb < NKB; ++kb) {
        const float* srow = ga.src[kb] + (long)m * ga.ld[kb];
        float4 f0 = *(const float4*)(srow + khalf);
        float4 f1 = *(const float4*)(srow + khalf + 4);
        float4 f2 = *(const float4*)(srow + 16 + khalf);
        float4 f3 = *(const float4*)(srow + 16 + khalf + 4);
        v16h a;
        a[0]  = (_Float16)f0.x; a[1]  = (_Float16)f0.y;
        a[2]  = (_Float16)f0.z; a[3]  = (_Float16)f0.w;
        a[4]  = (_Float16)f1.x; a[5]  = (_Float16)f1.y;
        a[6]  = (_Float16)f1.z; a[7]  = (_Float16)f1.w;
        a[8]  = (_Float16)f2.x; a[9]  = (_Float16)f2.y;
        a[10] = (_Float16)f2.z; a[11] = (_Float16)f2.w;
        a[12] = (_Float16)f3.x; a[13] = (_Float16)f3.y;
        a[14] = (_Float16)f3.z; a[15] = (_Float16)f3.w;

        v16h b = bfrag[kb * WPRT * 32];       // contiguous 32B per lane

        acc = __builtin_amdgcn_wmma_f32_16x16x32_f16(
            false, a, false, b, (short)0, acc, false, false);
    }

    int ncolg = (ntile << 4) + (lane & 15);
    float bias = ga.bias ? ga.bias[ncolg] : 0.0f;
    int mbase = (mtile << 4) + ((lane >> 4) << 3);   // D: VGPR r -> M = r (+8)
#pragma unroll
    for (int r = 0; r < 8; ++r)
        ga.out[(long)(mbase + r) * NCOLS + ncolg] = acc[r] + bias;
}

// H = (1 - sigmoid(Gz)) * tanh(Gh);  G is [N,128] = [Gz | Gh]
__global__ void hgate_kernel(const float* __restrict__ G, float* __restrict__ H, int n) {
    int i = blockIdx.x * blockDim.x + threadIdx.x;
    if (i >= n * 64) return;
    int m = i >> 6, f = i & 63;
    float z  = G[m * 128 + f];
    float ht = G[m * 128 + 64 + f];
    float zs = 1.0f / (1.0f + __expf(-z));
    H[i] = (1.0f - zs) * tanhf(ht);
}

// y = dis^2 * xw + gcn_b  (self-loop term; fully initializes y)
__global__ void yinit_kernel(const float* __restrict__ xw, const float* __restrict__ dis,
                             const float* __restrict__ gcn_b, float* __restrict__ y, int n) {
    int i = blockIdx.x * blockDim.x + threadIdx.x;
    if (i >= n * 64) return;
    int m = i >> 6, f = i & 63;
    float d = dis[m];
    y[i] = d * d * xw[i] + gcn_b[f];
}

// y[col,f] += dis[row]*dis[col] * xw[row,f]
__global__ void gcn_scatter_kernel(const float* __restrict__ xw, const float* __restrict__ dis,
                                   float* __restrict__ y,
                                   const int* __restrict__ row, const int* __restrict__ col,
                                   int E) {
    int gid = blockIdx.x * blockDim.x + threadIdx.x;
    int e = gid >> 6, f = gid & 63;
    if (e >= E) return;
    int r = row[e], c = col[e];
    float coeff = dis[r] * dis[c];
    atomicAdd(&y[c * 64 + f], coeff * xw[r * 64 + f]);
}

// Per-channel sum / sum-of-squares of relu(y) (grid-stride, then atomics)
__global__ void bnstat_kernel(const float* __restrict__ y, float* __restrict__ bnsum,
                              float* __restrict__ bnsq, int nrows) {
    int tid = threadIdx.x;
    int c = tid & 63;
    float s = 0.0f, q = 0.0f;
    for (int r = blockIdx.x * 4 + (tid >> 6); r < nrows; r += gridDim.x * 4) {
        float v = y[r * 64 + c];
        v = v > 0.0f ? v : 0.0f;
        s += v; q += v * v;
    }
    atomicAdd(&bnsum[c], s);
    atomicAdd(&bnsq[c], q);
}

// Fold BN affine + Linear into a[c] and scalar c:
//   out[m] = c + sum_f relu(y[m,f]) * a[f]
__global__ void coef_kernel(const float* __restrict__ bnsum, const float* __restrict__ bnsq,
                            const float* __restrict__ gamma, const float* __restrict__ beta,
                            const float* __restrict__ lin_w, const float* __restrict__ lin_b,
                            float* __restrict__ acoef, float* __restrict__ ccoef, int nrows) {
    __shared__ float red[64];
    int c = threadIdx.x;   // launched with 64 threads
    float invn = 1.0f / (float)nrows;
    float mean = bnsum[c] * invn;
    float var  = bnsq[c] * invn - mean * mean;
    float inv  = rsqrtf(var + BN_EPS);
    float g    = inv * gamma[c];
    acoef[c]   = g * lin_w[c];
    red[c]     = (beta[c] - mean * g) * lin_w[c];
    __syncthreads();
    for (int s = 32; s > 0; s >>= 1) {
        if (c < s) red[c] += red[c + s];
        __syncthreads();
    }
    if (c == 0) ccoef[0] = red[0] + lin_b[0];
}

__global__ void out_kernel(const float* __restrict__ y, const float* __restrict__ acoef,
                           const float* __restrict__ ccoef, float* __restrict__ out, int nrows) {
    int m = blockIdx.x * blockDim.x + threadIdx.x;
    if (m >= nrows) return;
    float s = ccoef[0];
    const float4* yr = (const float4*)(y + (long)m * 64);
    const float4* ar = (const float4*)acoef;
#pragma unroll
    for (int j = 0; j < 16; ++j) {
        float4 v = yr[j];
        float4 a = ar[j];
        float x0 = v.x > 0.0f ? v.x : 0.0f;
        float x1 = v.y > 0.0f ? v.y : 0.0f;
        float x2 = v.z > 0.0f ? v.z : 0.0f;
        float x3 = v.w > 0.0f ? v.w : 0.0f;
        s += x0 * a.x + x1 * a.y + x2 * a.z + x3 * a.w;
    }
    out[m] = s;
}

// ---------------------------------------------------------------------------
extern "C" void kernel_launch(void* const* d_in, const int* in_sizes, int n_in,
                              void* d_out, int out_size, void* d_ws, size_t ws_size,
                              hipStream_t stream) {
    const float* x      = (const float*)d_in[0];
    const int*   eidx   = (const int*)d_in[1];
    const float* ew     = (const float*)d_in[2];
    const float* Wz     = (const float*)d_in[3];
    const float* bz     = (const float*)d_in[4];
    // d_in[5], d_in[6] (Wr, br): provably unused since H0 == 0
    const float* Wh     = (const float*)d_in[7];
    const float* bh     = (const float*)d_in[8];
    const float* gcn_w  = (const float*)d_in[9];
    const float* gcn_b  = (const float*)d_in[10];
    const float* gamma  = (const float*)d_in[11];
    const float* beta   = (const float*)d_in[12];
    const float* lin_w  = (const float*)d_in[13];
    const float* lin_b  = (const float*)d_in[14];
    float* out = (float*)d_out;

    const int N = in_sizes[0] / N_FEAT;     // 50000
    const int E = in_sizes[2];              // 800000
    const int* row = eidx;
    const int* col = eidx + E;

    // --- workspace layout (bytes) ---
    char* ws = (char*)d_ws;
    const size_t oDegOut = 0;                          // N floats
    const size_t oDegIn  = 256 * 1024;
    const size_t oDcnt   = 512 * 1024;
    const size_t oDis    = 768 * 1024;
    const size_t oBnSum  = 1024 * 1024;                // 64 f
    const size_t oBnSq   = oBnSum + 1024;
    const size_t oAcoef  = oBnSum + 2048;
    const size_t oCcoef  = oBnSum + 3072;
    const size_t oBias   = oBnSum + 4096;              // 128 f
    const size_t oWgate  = oBnSum + 8192;              // 160*128 f16 = 40 KB
    const size_t oWgcn   = oWgate + 64 * 1024;         // 64*64 f16 = 8 KB
    const size_t oTx     = 2 * 1024 * 1024;            // 4 x (N*32 f) Tx buffers
    const size_t szTx    = (size_t)N * 32 * sizeof(float);       // 6.4 MB
    const size_t oTxO1 = oTx, oTxI1 = oTx + szTx;
    const size_t oTxO2 = oTx + 2 * szTx, oTxI2 = oTx + 3 * szTx;
    const size_t oG  = oTx + 4 * szTx;                 // N*128 f (Gz|Gh) = 25.6 MB
    const size_t oH  = oTxO1;                          // reuse Tx_o1+Tx_i1 (N*64 f)
    const size_t oXw = oTxO2;                          // reuse Tx_o2+Tx_i2 (N*64 f)
    const size_t oY  = oG;                             // reuse G after H computed

    float* deg_out = (float*)(ws + oDegOut);
    float* deg_in  = (float*)(ws + oDegIn);
    float* dcnt    = (float*)(ws + oDcnt);
    float* dis     = (float*)(ws + oDis);
    float* bnsum   = (float*)(ws + oBnSum);
    float* bnsq    = (float*)(ws + oBnSq);
    float* acoef   = (float*)(ws + oAcoef);
    float* ccoef   = (float*)(ws + oCcoef);
    float* biascat = (float*)(ws + oBias);
    _Float16* Wgate = (_Float16*)(ws + oWgate);
    _Float16* Wgcn  = (_Float16*)(ws + oWgcn);
    float* TxO1 = (float*)(ws + oTxO1);
    float* TxI1 = (float*)(ws + oTxI1);
    float* TxO2 = (float*)(ws + oTxO2);
    float* TxI2 = (float*)(ws + oTxI2);
    float* G  = (float*)(ws + oG);
    float* H  = (float*)(ws + oH);
    float* xw = (float*)(ws + oXw);
    float* y  = (float*)(ws + oY);

    // zero accumulation targets (degrees, bn sums, Tx buffers) every launch
    hipMemsetAsync(d_ws, 0, oG, stream);

    // weights -> f16 (+ fused bias)
    prep_weights_kernel<<<80, 256, 0, stream>>>(Wz, Wh, bz, bh, gcn_w, Wgate, Wgcn, biascat);

    // degrees
    degree_kernel<<<(E + 255) / 256, 256, 0, stream>>>(row, col, ew, deg_out, deg_in, dcnt, E);
    invert_kernel<<<(N + 255) / 256, 256, 0, stream>>>(deg_out, deg_in, dcnt, dis, N);

    // diffusion hops (shared across Z and H gates)
    int prop_blocks = (E * 32 + 255) / 256;
    prop_kernel<<<prop_blocks, 256, 0, stream>>>(x, x, TxO1, TxI1, row, col, ew, deg_out, deg_in, E);
    prop_kernel<<<prop_blocks, 256, 0, stream>>>(TxO1, TxI1, TxO2, TxI2, row, col, ew, deg_out, deg_in, E);

    // fused gate GEMM:  G[N,128] = [x|TxO1|TxI1|TxO2|TxI2] @ Wgate + [bz|bh]
    {
        GemmArgs ga;
        ga.src[0] = x;    ga.ld[0] = 32;
        ga.src[1] = TxO1; ga.ld[1] = 32;
        ga.src[2] = TxI1; ga.ld[2] = 32;
        ga.src[3] = TxO2; ga.ld[3] = 32;
        ga.src[4] = TxI2; ga.ld[4] = 32;
        ga.W = Wgate; ga.bias = biascat; ga.out = G; ga.nrows = N;
        gemm_wmma_kernel<5, 128><<<N / 16, 256, 0, stream>>>(ga);
    }

    // H = (1 - sigmoid(Gz)) * tanh(Gh)
    hgate_kernel<<<(N * 64 + 255) / 256, 256, 0, stream>>>(G, H, N);

    // xw = H @ gcn_w
    {
        GemmArgs ga;
        ga.src[0] = H;      ga.ld[0] = 64;
        ga.src[1] = H + 32; ga.ld[1] = 64;
        ga.src[2] = H; ga.ld[2] = 64;
        ga.src[3] = H; ga.ld[3] = 64;
        ga.src[4] = H; ga.ld[4] = 64;
        ga.W = Wgcn; ga.bias = nullptr; ga.out = xw; ga.nrows = N;
        int ntiles = N / 16;
        gemm_wmma_kernel<2, 64><<<(ntiles + 1) / 2, 256, 0, stream>>>(ga);
    }

    // GCN aggregation with symmetric norm + self loops
    yinit_kernel<<<(N * 64 + 255) / 256, 256, 0, stream>>>(xw, dis, gcn_b, y, N);
    gcn_scatter_kernel<<<(E * 64 + 255) / 256, 256, 0, stream>>>(xw, dis, y, row, col, E);

    // ReLU -> BatchNorm (batch stats) -> Linear, folded
    bnstat_kernel<<<256, 256, 0, stream>>>(y, bnsum, bnsq, N);
    coef_kernel<<<1, 64, 0, stream>>>(bnsum, bnsq, gamma, beta, lin_w, lin_b, acoef, ccoef, N);
    out_kernel<<<(N + 255) / 256, 256, 0, stream>>>(y, acoef, ccoef, out, N);
}
